// SplittingMLPNetwork_78778290143359
// MI455X (gfx1250) — compile-verified, matrix operand order checked by reading, not run on hardware
//
#include <hip/hip_runtime.h>
#include <math.h>

#define N_SAMP 524288
#define T_TASK 10
#define F_IN   128
#define F_HID  64
#define F_OUT  64

typedef __attribute__((ext_vector_type(16))) __bf16 v16bf;
typedef __attribute__((ext_vector_type(8)))  float  v8f;

union FragU { uint4 u[2]; v16bf v; };

__device__ __forceinline__ unsigned int f2bf(float f) {
  unsigned int u = __float_as_uint(f);
  return (u + 0x7FFFu + ((u >> 16) & 1u)) >> 16;   // RNE f32 -> bf16
}

// Same-wave LDS phase fence: LDS pipe is in-order per wave; the s_wait_dscnt
// plus the "memory" clobber stops both HW surprises and compiler reordering
// of the type-punned shared-memory accesses across phases.
__device__ __forceinline__ void wave_lds_fence() {
  asm volatile("s_wait_dscnt 0" ::: "memory");
}

// ---------------- counting sort by task ----------------
__global__ void init_counts(int* hist, int* cursors) {
  int t = threadIdx.x;
  if (t < T_TASK) { hist[t] = 0; cursors[t] = 0; }
}

__global__ void hist_kernel(const int* __restrict__ task, int* __restrict__ hist) {
  int i = blockIdx.x * blockDim.x + threadIdx.x;
  if (i < N_SAMP) atomicAdd(&hist[task[i]], 1);
}

__global__ void scan_kernel(const int* __restrict__ hist, int* __restrict__ segstart) {
  if (threadIdx.x == 0) {
    int run = 0;
    for (int t = 0; t < T_TASK; ++t) { segstart[t] = run; run += hist[t]; }
  }
}

__global__ void scatter_kernel(const int* __restrict__ task,
                               const int* __restrict__ segstart,
                               int* __restrict__ cursors,
                               int* __restrict__ perm) {
  int i = blockIdx.x * blockDim.x + threadIdx.x;
  if (i < N_SAMP) {
    int t = task[i];
    int p = atomicAdd(&cursors[t], 1);
    perm[segstart[t] + p] = i;
  }
}

// ---------------- fused routed MLP ----------------
// Stage a [fi x fo] f32 weight matrix into LDS as bf16, pre-swizzled into
// B-fragment lane layout for v_wmma_f32_16x16x32_bf16:
//   fragment f = ks*4 + nt holds K in [ks*32, ks*32+32), N in [nt*16, nt*16+16)
//   lane L (n = nt*16 + (L&15), half = L>>4) stores its 16 bf16 contiguously.
__device__ __forceinline__ void stageW(unsigned short* dst, const float* src,
                                       int ksteps, int fo, int tid) {
  const int total = ksteps * 4 * 512;
  for (int idx = tid; idx < total; idx += 256) {
    int f    = idx >> 9;
    int rem  = idx & 511;
    int L    = rem >> 4;
    int s    = rem & 15;
    int nt   = f & 3;
    int ks   = f >> 2;
    int half = L >> 4;
    int j    = s >> 1;
    int krel = (j < 4) ? (half * 8 + 2 * j + (s & 1))
                       : (16 + half * 8 + 2 * (j - 4) + (s & 1));
    int k = ks * 32 + krel;
    int n = nt * 16 + (L & 15);
    dst[idx] = (unsigned short)f2bf(src[k * fo + n]);
  }
}

__global__ __launch_bounds__(256)
void mlp_kernel(const float* __restrict__ X,
                const int*   __restrict__ perm,
                const int*   __restrict__ hist,
                const int*   __restrict__ segstart,
                const int*   __restrict__ maps,
                const float* __restrict__ W0, const float* __restrict__ b0,
                const float* __restrict__ W1, const float* __restrict__ b1,
                const float* __restrict__ W2, const float* __restrict__ b2,
                float* __restrict__ out) {
  const int task  = blockIdx.y;
  const int count = hist[task];
  const int tiles = (count + 15) >> 4;
  if ((int)blockIdx.x * 8 >= tiles) return;
  const int segb = segstart[task];

  const int c0 = maps[0 * T_TASK + task];
  const int c1 = maps[1 * T_TASK + task];
  const int c2 = maps[2 * T_TASK + task];

  __shared__ __align__(16) unsigned short w0s[16 * 512];   // 16 KB
  __shared__ __align__(16) unsigned short w1s[8 * 512];    //  8 KB
  __shared__ __align__(16) unsigned short w2s[8 * 512];    //  8 KB
  __shared__ float b0s[64], b1s[64], b2s[64];
  __shared__ __align__(16) unsigned short xs[8][16 * 128]; // 32 KB (per-wave X tiles)
  __shared__ __align__(16) unsigned short hs[8][16 * 64];  // 16 KB (per-wave H tiles)
  __shared__ int rowg[8][16];

  const int tid = threadIdx.x;
  stageW(w0s, W0 + c0 * (F_IN  * F_HID), 4, F_HID, tid);
  stageW(w1s, W1 + c1 * (F_HID * F_HID), 2, F_HID, tid);
  stageW(w2s, W2 + c2 * (F_HID * F_OUT), 2, F_OUT, tid);
  if (tid < 64) {
    b0s[tid] = b0[c0 * 64 + tid];
    b1s[tid] = b1[c1 * 64 + tid];
    b2s[tid] = b2[c2 * 64 + tid];
  }
  __syncthreads();

  const int wave = tid >> 5;
  const int lane = tid & 31;
  const int m    = lane & 15;
  const int half = lane >> 4;

  unsigned short* xw = xs[wave];
  unsigned short* hw = hs[wave];
  int* rg = rowg[wave];

  for (int tile = blockIdx.x * 8 + wave; tile < tiles; tile += 8 * (int)gridDim.x) {
    // ---- gather 16 rows, convert to bf16, stage row-major in LDS ----
    int r  = tile * 16 + m;
    int rc = (r < count) ? r : (count - 1);
    int g  = perm[segb + rc];
    if (half == 0) rg[m] = (r < count) ? g : -1;
    const float4* rp = (const float4*)(X + (size_t)g * F_IN + half * 64);
#pragma unroll
    for (int c = 0; c < 8; ++c) {
      float4 f0 = rp[c * 2];
      float4 f1 = rp[c * 2 + 1];
      uint4 pk;
      pk.x = f2bf(f0.x) | (f2bf(f0.y) << 16);
      pk.y = f2bf(f0.z) | (f2bf(f0.w) << 16);
      pk.z = f2bf(f1.x) | (f2bf(f1.y) << 16);
      pk.w = f2bf(f1.z) | (f2bf(f1.w) << 16);
      *(uint4*)&xw[m * 128 + half * 64 + c * 8] = pk;
    }
    wave_lds_fence();

    v8f acc[4];

    // ---- layer 0: [16x128] @ [128x64], bias in C, tanh ----
    FragU a0[4];
#pragma unroll
    for (int ks = 0; ks < 4; ++ks) {
      int kb = ks * 32 + half * 8;
      a0[ks].u[0] = *(const uint4*)&xw[m * 128 + kb];
      a0[ks].u[1] = *(const uint4*)&xw[m * 128 + kb + 16];
    }
#pragma unroll
    for (int nt = 0; nt < 4; ++nt) {
      float bv = b0s[nt * 16 + m];
#pragma unroll
      for (int v = 0; v < 8; ++v) acc[nt][v] = bv;
#pragma unroll
      for (int ks = 0; ks < 4; ++ks) {
        FragU bf;
        const unsigned short* wp = &w0s[(ks * 4 + nt) * 512 + lane * 16];
        bf.u[0] = *(const uint4*)wp;
        bf.u[1] = *(const uint4*)(wp + 8);
        acc[nt] = __builtin_amdgcn_wmma_f32_16x16x32_bf16(
            false, a0[ks].v, false, bf.v, (short)0, acc[nt], false, false);
      }
    }
    wave_lds_fence();
#pragma unroll
    for (int nt = 0; nt < 4; ++nt)
#pragma unroll
      for (int v = 0; v < 8; ++v)
        hw[(v + half * 8) * 64 + nt * 16 + m] = (unsigned short)f2bf(tanhf(acc[nt][v]));
    wave_lds_fence();

    // ---- layer 1: [16x64] @ [64x64], tanh ----
    FragU a1[2];
#pragma unroll
    for (int ks = 0; ks < 2; ++ks) {
      int kb = ks * 32 + half * 8;
      a1[ks].u[0] = *(const uint4*)&hw[m * 64 + kb];
      a1[ks].u[1] = *(const uint4*)&hw[m * 64 + kb + 16];
    }
#pragma unroll
    for (int nt = 0; nt < 4; ++nt) {
      float bv = b1s[nt * 16 + m];
#pragma unroll
      for (int v = 0; v < 8; ++v) acc[nt][v] = bv;
#pragma unroll
      for (int ks = 0; ks < 2; ++ks) {
        FragU bf;
        const unsigned short* wp = &w1s[(ks * 4 + nt) * 512 + lane * 16];
        bf.u[0] = *(const uint4*)wp;
        bf.u[1] = *(const uint4*)(wp + 8);
        acc[nt] = __builtin_amdgcn_wmma_f32_16x16x32_bf16(
            false, a1[ks].v, false, bf.v, (short)0, acc[nt], false, false);
      }
    }
    wave_lds_fence();
#pragma unroll
    for (int nt = 0; nt < 4; ++nt)
#pragma unroll
      for (int v = 0; v < 8; ++v)
        hw[(v + half * 8) * 64 + nt * 16 + m] = (unsigned short)f2bf(tanhf(acc[nt][v]));
    wave_lds_fence();

    // ---- layer 2: [16x64] @ [64x64], no activation ----
    FragU a2[2];
#pragma unroll
    for (int ks = 0; ks < 2; ++ks) {
      int kb = ks * 32 + half * 8;
      a2[ks].u[0] = *(const uint4*)&hw[m * 64 + kb];
      a2[ks].u[1] = *(const uint4*)&hw[m * 64 + kb + 16];
    }
#pragma unroll
    for (int nt = 0; nt < 4; ++nt) {
      float bv = b2s[nt * 16 + m];
#pragma unroll
      for (int v = 0; v < 8; ++v) acc[nt][v] = bv;
#pragma unroll
      for (int ks = 0; ks < 2; ++ks) {
        FragU bf;
        const unsigned short* wp = &w2s[(ks * 4 + nt) * 512 + lane * 16];
        bf.u[0] = *(const uint4*)wp;
        bf.u[1] = *(const uint4*)(wp + 8);
        acc[nt] = __builtin_amdgcn_wmma_f32_16x16x32_bf16(
            false, a2[ks].v, false, bf.v, (short)0, acc[nt], false, false);
      }
    }
    wave_lds_fence();

    // ---- scatter f32 output, masking padded rows ----
#pragma unroll
    for (int v = 0; v < 8; ++v) {
      int gr = rg[v + half * 8];
      if (gr >= 0) {
        float* op = out + (size_t)gr * F_OUT + m;
#pragma unroll
        for (int nt = 0; nt < 4; ++nt) op[nt * 16] = acc[nt][v];
      }
    }
  }
}

extern "C" void kernel_launch(void* const* d_in, const int* in_sizes, int n_in,
                              void* d_out, int out_size, void* d_ws, size_t ws_size,
                              hipStream_t stream) {
  const float* X    = (const float*)d_in[0];
  const int*   task = (const int*)d_in[1];
  const int*   maps = (const int*)d_in[2];
  const float* W0   = (const float*)d_in[3];
  const float* b0   = (const float*)d_in[4];
  const float* W1   = (const float*)d_in[5];
  const float* b1   = (const float*)d_in[6];
  const float* W2   = (const float*)d_in[7];
  const float* b2   = (const float*)d_in[8];
  float* out = (float*)d_out;

  int* ws       = (int*)d_ws;
  int* perm     = ws;                  // N ints
  int* hist     = ws + N_SAMP;         // 10 (padded 16)
  int* cursors  = ws + N_SAMP + 16;    // 10 (padded 16)
  int* segstart = ws + N_SAMP + 32;    // 10

  init_counts<<<1, 32, 0, stream>>>(hist, cursors);
  hist_kernel<<<N_SAMP / 256, 256, 0, stream>>>(task, hist);
  scan_kernel<<<1, 1, 0, stream>>>(hist, segstart);
  scatter_kernel<<<N_SAMP / 256, 256, 0, stream>>>(task, segstart, cursors, perm);
  mlp_kernel<<<dim3(512, T_TASK), 256, 0, stream>>>(
      X, perm, hist, segstart, maps, W0, b0, W1, b1, W2, b2, out);
}
